// SOGC_DSBN_47313359733019
// MI455X (gfx1250) — compile-verified, matrix operand order checked by reading, not run on hardware
//
#include <hip/hip_runtime.h>
#include <hip/hip_bf16.h>

#define BATCH 2048
#define CCH   62
#define FIN   400
#define KPAD  416           // FIN padded to multiple of 32
#define FBN   64
#define FOUT  128
#define MROWS (BATCH*CCH)   // 126976 = 7936 * 16
#define FFLAT (CCH*FBN)     // 3968
#define KTOP  10
#define BN_EPS 1e-5f
#define SAS   72            // LDS row stride (bf16): 144B, 16B-aligned, bank-skewed
#define SHS   72
#define WSL   40            // weight-slice LDS row stride (bf16): 80B, 16B-aligned,
                            // 20 dwords -> 16 rows hit 16 distinct banks
#define SLICE (192*WSL)     // one K-slice: 192 rows x 32 K (padded to 40)

typedef __attribute__((ext_vector_type(16))) __bf16   bf16x16;
typedef __attribute__((ext_vector_type(2)))  __bf16   bf16x2;
typedef __attribute__((ext_vector_type(8)))  float    f32x8;
typedef __attribute__((ext_vector_type(4)))  float    f32x4;
typedef __attribute__((ext_vector_type(8)))  unsigned u32x8;
typedef __attribute__((ext_vector_type(4)))  unsigned u32x4;

static __device__ __forceinline__ __bf16 f2bf(float f) {
  union { float f; unsigned u; } v; v.f = f;
  unsigned r = v.u + 0x7fffu + ((v.u >> 16) & 1u);   // round-to-nearest-even
  unsigned short h = (unsigned short)(r >> 16);
  __bf16 b; __builtin_memcpy(&b, &h, sizeof(b));
  return b;
}

// pack two f32 into one dword of two bf16, lo in [15:0]
#if __has_builtin(__builtin_amdgcn_cvt_pk_bf16_f32)
static __device__ __forceinline__ unsigned pack_bf2(float lo, float hi) {
  bf16x2 p = __builtin_amdgcn_cvt_pk_bf16_f32(lo, hi);
  return __builtin_bit_cast(unsigned, p);
}
#else
static __device__ __forceinline__ unsigned pack_bf2(float lo, float hi) {
  union { float f; unsigned u; } a, c; a.f = lo; c.f = hi;
  unsigned ra = a.u + 0x7fffu + ((a.u >> 16) & 1u);
  unsigned rc = c.u + 0x7fffu + ((c.u >> 16) & 1u);
  return (ra >> 16) | (rc & 0xffff0000u);
}
#endif

static __device__ __forceinline__ f32x8 wmma_bf16(bf16x16 a, bf16x16 b, f32x8 c) {
  return __builtin_amdgcn_wmma_f32_16x16x32_bf16(false, a, false, b, (short)0, c, false, false);
}

// ---- fragment loaders -------------------------------------------------------
// 16-bit fragment layout (ISA 7.12.2): lane's VGPR v holds K pairs {2v,2v+1},
// v>=4 adds +16; lanes 16..31 add +8.  => two contiguous 8-elem runs per lane.

static __device__ __forceinline__ bf16x16 ld_bf16_row(const __bf16* base, int stride,
                                                      int row, int k0) {
  const int lane = threadIdx.x & 31;
  const int kb = k0 + ((lane >> 4) << 3);
  const unsigned* p = (const unsigned*)(base + (size_t)row * stride + kb);
  u32x4 lo = *(const u32x4*)p;        // K = kb .. kb+7
  u32x4 hi = *(const u32x4*)(p + 8);  // K = kb+16 .. kb+23
  u32x8 u;
  u[0] = lo[0]; u[1] = lo[1]; u[2] = lo[2]; u[3] = lo[3];
  u[4] = hi[0]; u[5] = hi[1]; u[6] = hi[2]; u[7] = hi[3];
  return __builtin_bit_cast(bf16x16, u);
}

static __device__ __forceinline__ bf16x16 ld_f32_row_full(const float* p, int k0) {
  const int lane = threadIdx.x & 31;
  const int kb = k0 + ((lane >> 4) << 3);
  f32x4 a0 = *(const f32x4*)(p + kb);
  f32x4 a1 = *(const f32x4*)(p + kb + 4);
  f32x4 a2 = *(const f32x4*)(p + kb + 16);
  f32x4 a3 = *(const f32x4*)(p + kb + 20);
  u32x8 u;
  u[0] = pack_bf2(a0[0], a0[1]); u[1] = pack_bf2(a0[2], a0[3]);
  u[2] = pack_bf2(a1[0], a1[1]); u[3] = pack_bf2(a1[2], a1[3]);
  u[4] = pack_bf2(a2[0], a2[1]); u[5] = pack_bf2(a2[2], a2[3]);
  u[6] = pack_bf2(a3[0], a3[1]); u[7] = pack_bf2(a3[2], a3[3]);
  return __builtin_bit_cast(bf16x16, u);
}

// tail (k0=384): K kb..kb+7 in range, kb+16..kb+23 zero
static __device__ __forceinline__ bf16x16 ld_f32_row_tail(const float* p, int k0) {
  const int lane = threadIdx.x & 31;
  const int kb = k0 + ((lane >> 4) << 3);
  f32x4 a0 = *(const f32x4*)(p + kb);
  f32x4 a1 = *(const f32x4*)(p + kb + 4);
  u32x8 u;
  u[0] = pack_bf2(a0[0], a0[1]); u[1] = pack_bf2(a0[2], a0[3]);
  u[2] = pack_bf2(a1[0], a1[1]); u[3] = pack_bf2(a1[2], a1[3]);
  u[4] = 0u; u[5] = 0u; u[6] = 0u; u[7] = 0u;
  return __builtin_bit_cast(bf16x16, u);
}

// ---- wave32 reductions ------------------------------------------------------
static __device__ __forceinline__ float wave_max_f(float v) {
#pragma unroll
  for (int o = 16; o > 0; o >>= 1) v = fmaxf(v, __shfl_xor(v, o, 32));
  return v;
}
static __device__ __forceinline__ float wave_sum_f(float v) {
#pragma unroll
  for (int o = 16; o > 0; o >>= 1) v += __shfl_xor(v, o, 32);
  return v;
}
static __device__ __forceinline__ int wave_min_i(int v) {
#pragma unroll
  for (int o = 16; o > 0; o >>= 1) { int u = __shfl_xor(v, o, 32); v = (u < v) ? u : v; }
  return v;
}

// ---- weight-slice staging (global -> regs -> LDS) ---------------------------
// slice = wbf[0:192][k0:k0+32]; 768 16B chunks; 256 threads x 3 chunks.
static __device__ __forceinline__ void fetch_slice(const __bf16* __restrict__ wbf, int k0,
                                                   int tid, u32x4& t0, u32x4& t1, u32x4& t2) {
  const int c0 = tid, c1 = tid + 256, c2 = tid + 512;
  t0 = *(const u32x4*)(wbf + (c0 >> 2) * KPAD + k0 + (c0 & 3) * 8);
  t1 = *(const u32x4*)(wbf + (c1 >> 2) * KPAD + k0 + (c1 & 3) * 8);
  t2 = *(const u32x4*)(wbf + (c2 >> 2) * KPAD + k0 + (c2 & 3) * 8);
}
static __device__ __forceinline__ void store_slice(__bf16* sw, int tid,
                                                   const u32x4& t0, const u32x4& t1,
                                                   const u32x4& t2) {
  const int c0 = tid, c1 = tid + 256, c2 = tid + 512;
  *(u32x4*)(sw + (c0 >> 2) * WSL + (c0 & 3) * 8) = t0;
  *(u32x4*)(sw + (c1 >> 2) * WSL + (c1 & 3) * 8) = t1;
  *(u32x4*)(sw + (c2 >> 2) * WSL + (c2 & 3) * 8) = t2;
}

// ---------------------------------------------------------------------------
// Kernel 0: weights -> bf16 panel wbf (192 x 416), zero K-padded.
//   rows   0..63 : bnW[n][k]             (B[k][n] = bnW[n][k], xa GEMM)
//   rows 64..191 : gcnW[k][n] transposed (B[k][n] = gcnWt[n][k], h GEMM)
// ---------------------------------------------------------------------------
__global__ __launch_bounds__(256)
void k_prep_weights(const float* __restrict__ bnW, const float* __restrict__ gcnW,
                    __bf16* __restrict__ wbf) {
  const int idx = blockIdx.x * 256 + threadIdx.x;
  if (idx >= 192 * KPAD) return;
  const int row = idx / KPAD, k = idx % KPAD;
  float v = 0.f;
  if (k < FIN)
    v = (row < FBN) ? bnW[row * FIN + k] : gcnW[k * FOUT + (row - FBN)];
  wbf[idx] = f2bf(v);
}

// ---------------------------------------------------------------------------
// Kernel 1: fused x@bnWᵀ (+bias) and x@gcnW — one pass over x (203 MB).
// Wave = 16(M) x 192(N) strip, 12 f32x8 accumulators. The 12 B fragments per
// k-step are identical across all 8 waves, so the block stages each 12 KB
// weight K-slice into LDS (double-buffered); the next slice is fetched into
// registers BEFORE the 12 WMMAs so global latency hides behind matrix work,
// and B operands become short-latency ds_load_b128 pairs (8x less L2 traffic).
// ---------------------------------------------------------------------------
__global__ __launch_bounds__(256)
void k_fused_gemms(const float* __restrict__ x,
                   const __bf16* __restrict__ wbf,
                   const float* __restrict__ bnB,
                   float* __restrict__ xa_ws,      // (MROWS, 64)
                   float* __restrict__ h_ws) {     // (MROWS, 128)
  const int tid  = threadIdx.x;
  const int lane = tid & 31;
  const int wave = tid >> 5;
  const int lr   = lane & 15;
  const int hi8  = (lane >> 4) << 3;
  const int tm   = blockIdx.x * 8 + wave;   // 16-row tile (0..7935)
  const int m    = tm * 16 + lr;
  const float* xrow = x + (size_t)m * FIN;

  __shared__ __bf16 wS[2 * SLICE];          // 2 x 15 KB weight K-slices

  f32x8 acc[12];
#pragma unroll
  for (int s = 0; s < 12; ++s) acc[s] = f32x8{};

  { // prologue: stage slice 0
    u32x4 t0, t1, t2;
    fetch_slice(wbf, 0, tid, t0, t1, t2);
    store_slice(wS, tid, t0, t1, t2);
  }
  __syncthreads();

  for (int step = 0; step < 12; ++step) {
    const int k0 = step * 32;
    const __bf16* sw = wS + (step & 1) * SLICE;
    u32x4 t0, t1, t2;
    fetch_slice(wbf, k0 + 32, tid, t0, t1, t2);   // next slice -> regs (overlaps WMMA)
    if (k0 + 64 < FIN)
      __builtin_prefetch(xrow + k0 + 64, 0, 1);   // global_prefetch_b8 for x
    bf16x16 a = ld_f32_row_full(xrow, k0);
#pragma unroll
    for (int g = 0; g < 3; ++g) {
      bf16x16 b0 = ld_bf16_row(sw, WSL, (g * 4 + 0) * 16 + lr, 0);
      bf16x16 b1 = ld_bf16_row(sw, WSL, (g * 4 + 1) * 16 + lr, 0);
      bf16x16 b2 = ld_bf16_row(sw, WSL, (g * 4 + 2) * 16 + lr, 0);
      bf16x16 b3 = ld_bf16_row(sw, WSL, (g * 4 + 3) * 16 + lr, 0);
      acc[g * 4 + 0] = wmma_bf16(a, b0, acc[g * 4 + 0]);
      acc[g * 4 + 1] = wmma_bf16(a, b1, acc[g * 4 + 1]);
      acc[g * 4 + 2] = wmma_bf16(a, b2, acc[g * 4 + 2]);
      acc[g * 4 + 3] = wmma_bf16(a, b3, acc[g * 4 + 3]);
    }
    store_slice(wS + ((step + 1) & 1) * SLICE, tid, t0, t1, t2);
    __syncthreads();
  }
  { // K tail 384..399 (weight panel zero-padded); slice 12 lives in buffer 0
    const __bf16* sw = wS + (12 & 1) * SLICE;
    bf16x16 a = ld_f32_row_tail(xrow, 384);
#pragma unroll
    for (int g = 0; g < 3; ++g) {
      bf16x16 b0 = ld_bf16_row(sw, WSL, (g * 4 + 0) * 16 + lr, 0);
      bf16x16 b1 = ld_bf16_row(sw, WSL, (g * 4 + 1) * 16 + lr, 0);
      bf16x16 b2 = ld_bf16_row(sw, WSL, (g * 4 + 2) * 16 + lr, 0);
      bf16x16 b3 = ld_bf16_row(sw, WSL, (g * 4 + 3) * 16 + lr, 0);
      acc[g * 4 + 0] = wmma_bf16(a, b0, acc[g * 4 + 0]);
      acc[g * 4 + 1] = wmma_bf16(a, b1, acc[g * 4 + 1]);
      acc[g * 4 + 2] = wmma_bf16(a, b2, acc[g * 4 + 2]);
      acc[g * 4 + 3] = wmma_bf16(a, b3, acc[g * 4 + 3]);
    }
  }

  const float bb0 = bnB[lr], bb1 = bnB[16 + lr], bb2 = bnB[32 + lr], bb3 = bnB[48 + lr];
#pragma unroll
  for (int r = 0; r < 8; ++r) {
    const size_t row = (size_t)tm * 16 + r + hi8;
    float* xo = xa_ws + row * FBN + lr;
    xo[ 0] = acc[0][r] + bb0;
    xo[16] = acc[1][r] + bb1;
    xo[32] = acc[2][r] + bb2;
    xo[48] = acc[3][r] + bb3;
    float* ho = h_ws + row * FOUT + lr;
#pragma unroll
    for (int s = 0; s < 8; ++s) ho[s * 16] = acc[4 + s][r];
  }
}

// ---------------------------------------------------------------------------
// Kernel 2: BN statistics over B=2048 for 3968 flat features (coalesced).
// ---------------------------------------------------------------------------
__global__ __launch_bounds__(256)
void k_bn_stats(const float* __restrict__ xa_ws,
                float* __restrict__ mean_ws, float* __restrict__ rstd_ws) {
  const int f = blockIdx.x * 256 + threadIdx.x;
  if (f >= FFLAT) return;
  float s = 0.f, ss = 0.f;
  for (int b = 0; b < BATCH; ++b) {
    float v = xa_ws[(size_t)b * FFLAT + f];
    s += v; ss += v * v;
  }
  const float mu  = s * (1.0f / BATCH);
  const float var = ss * (1.0f / BATCH) - mu * mu;   // biased var (BatchNorm)
  mean_ws[f] = mu;
  rstd_ws[f] = rsqrtf(var + BN_EPS);
}

// ---------------------------------------------------------------------------
// Kernel 3: per-batch graph stage, one block (8 wave32) per batch element.
// ---------------------------------------------------------------------------
__global__ __launch_bounds__(256)
void k_graph(const float* __restrict__ xa_ws, const float* __restrict__ h_ws,
             const float* __restrict__ mean_ws, const float* __restrict__ rstd_ws,
             const float* __restrict__ gamma, const float* __restrict__ beta,
             const float* __restrict__ gcnB, float* __restrict__ out) {
  const int b    = blockIdx.x;
  const int tid  = threadIdx.x;
  const int lane = tid & 31;
  const int wave = tid >> 5;
  const int lr   = lane & 15;
  const int hi8  = (lane >> 4) << 3;

  __shared__ __bf16 sA[64 * SAS];     // tanh(BN(xa)); later normalized adj
  __shared__ __bf16 sHt[FOUT * SHS];  // H transposed: sHt[o*SHS + r] = h[r][o]
  __shared__ float  sS[64 * 64];      // similarity / adjacency (f32)
  __shared__ float  sd[64];           // D^{-1/2}

  // ---- BN + tanh -> bf16 (rows 62,63 zero) ----
  for (int idx = tid; idx < 64 * 64; idx += 256) {
    const int r = idx >> 6, g = idx & 63;
    float v = 0.f;
    if (r < CCH) {
      const int f = r * FBN + g;
      const float xv = xa_ws[((size_t)b * CCH + r) * FBN + g];
      v = tanhf((xv - mean_ws[f]) * rstd_ws[f] * gamma[f] + beta[f]);
    }
    sA[r * SAS + g] = f2bf(v);
  }
  // ---- H -> transposed bf16 tile (cols r>=62 zero) ----
  for (int idx = tid; idx < 64 * FOUT; idx += 256) {
    const int r = idx >> 7, o = idx & 127;   // coalesced read of h_ws
    const float v = (r < CCH) ? h_ws[((size_t)b * CCH + r) * FOUT + o] : 0.f;
    sHt[o * SHS + r] = f2bf(v);
  }
  __syncthreads();

  // ---- S = A @ Aᵀ : 16 tiles of 16x16, 2 per wave, K = 64 ----
#pragma unroll
  for (int t = 0; t < 2; ++t) {
    const int tile = wave * 2 + t;
    const int tmr = (tile >> 2) * 16, tnc = (tile & 3) * 16;
    f32x8 acc = {};
#pragma unroll
    for (int k0 = 0; k0 < 64; k0 += 32) {
      bf16x16 a  = ld_bf16_row(sA, SAS, tmr + lr, k0);
      bf16x16 bt = ld_bf16_row(sA, SAS, tnc + lr, k0);  // B[k][n] = A[n][k]
      acc = wmma_bf16(a, bt, acc);
    }
#pragma unroll
    for (int r = 0; r < 8; ++r)
      sS[(tmr + r + hi8) * 64 + tnc + lr] = acc[r];
  }
  __syncthreads();

  // ---- per-row: softmax over 62 cols, top-10 mask, self-loop, degree ----
  for (int r = wave; r < CCH; r += 8) {
    const int c0 = lane, c1 = lane + 32;
    const bool ok0 = (c0 < CCH), ok1 = (c1 < CCH);
    const float s0 = ok0 ? sS[r * 64 + c0] : -1e30f;
    const float s1 = ok1 ? sS[r * 64 + c1] : -1e30f;
    const float mx = wave_max_f(fmaxf(s0, s1));
    const float e0 = ok0 ? __expf(s0 - mx) : 0.f;
    const float e1 = ok1 ? __expf(s1 - mx) : 0.f;
    const float inv = 1.0f / wave_sum_f(e0 + e1);
    const float p0 = e0 * inv, p1 = e1 * inv;

    bool keep0 = false, keep1 = false;
#pragma unroll
    for (int it = 0; it < KTOP; ++it) {
      const float q0 = (ok0 && !keep0) ? p0 : -1.f;
      const float q1 = (ok1 && !keep1) ? p1 : -1.f;
      const float mv = wave_max_f(fmaxf(q0, q1));
      int cand = 0x7fffffff;
      if (q0 == mv) cand = c0;
      if (q1 == mv && c1 < cand) cand = c1;
      cand = wave_min_i(cand);
      if (cand == c0) keep0 = true;
      else if (cand == c1) keep1 = true;
    }
    float w0 = keep0 ? p0 : 0.f;
    float w1 = keep1 ? p1 : 0.f;
    if (c0 == r) w0 = 1.0f;      // self loop overrides mask (as in reference)
    if (c1 == r) w1 = 1.0f;
    if (!ok0) w0 = 0.f;
    if (!ok1) w1 = 0.f;
    sS[r * 64 + c0] = w0;
    sS[r * 64 + c1] = w1;
    const float dsum = wave_sum_f(w0 + w1);
    if (lane == 0) sd[r] = rsqrtf(fmaxf(dsum, 1.0f));
  }
  __syncthreads();

  // ---- normalized adjacency -> bf16 into sA (reused) ----
  for (int idx = tid; idx < 64 * 64; idx += 256) {
    const int r = idx >> 6, c = idx & 63;
    const float v = (r < CCH && c < CCH) ? sS[idx] * sd[r] * sd[c] : 0.f;
    sA[r * SAS + c] = f2bf(v);
  }
  __syncthreads();

  // ---- out = adj @ h : 4x8 tiles of 16x16, 4 per wave, K = 64 ----
#pragma unroll
  for (int t = 0; t < 4; ++t) {
    const int tile = wave * 4 + t;
    const int tmr = (tile >> 3) * 16, tnc = (tile & 7) * 16;
    f32x8 acc = {};
#pragma unroll
    for (int k0 = 0; k0 < 64; k0 += 32) {
      bf16x16 a  = ld_bf16_row(sA, SAS, tmr + lr, k0);
      bf16x16 bt = ld_bf16_row(sHt, SHS, tnc + lr, k0);  // B[k][n] = sHt[n][k]
      acc = wmma_bf16(a, bt, acc);
    }
#pragma unroll
    for (int r = 0; r < 8; ++r) {
      const int row = tmr + r + hi8;
      if (row < CCH) {
        const int n = tnc + lr;
        const float v = acc[r] + gcnB[n];
        out[((size_t)b * CCH + row) * FOUT + n] = fmaxf(v, 0.f);
      }
    }
  }
}

extern "C" void kernel_launch(void* const* d_in, const int* in_sizes, int n_in,
                              void* d_out, int out_size, void* d_ws, size_t ws_size,
                              hipStream_t stream) {
  const float* x    = (const float*)d_in[0];  // (2048, 62, 400)
  const float* bnW  = (const float*)d_in[1];  // (64, 400)
  const float* bnB  = (const float*)d_in[2];  // (64,)
  const float* gmma = (const float*)d_in[3];  // (3968,)
  const float* beta = (const float*)d_in[4];  // (3968,)
  const float* gcnW = (const float*)d_in[5];  // (400, 128)
  const float* gcnB = (const float*)d_in[6];  // (128,)
  // d_in[7] = topk (==10, hardcoded as KTOP)
  float* outp = (float*)d_out;                // (2048, 62, 128)

  float*  xa_ws   = (float*)d_ws;                       // MROWS*64 f32
  float*  h_ws    = xa_ws + (size_t)MROWS * FBN;        // MROWS*128 f32
  float*  mean_ws = h_ws + (size_t)MROWS * FOUT;        // 3968 f32
  float*  rstd_ws = mean_ws + FFLAT;                    // 3968 f32
  __bf16* wbf     = (__bf16*)(rstd_ws + FFLAT);         // 192*416 bf16 (16B aligned)

  k_prep_weights<<<(192 * KPAD + 255) / 256, 256, 0, stream>>>(bnW, gcnW, wbf);
  k_fused_gemms<<<MROWS / 16 / 8, 256, 0, stream>>>(x, wbf, bnB, xa_ws, h_ws);
  k_bn_stats<<<(FFLAT + 255) / 256, 256, 0, stream>>>(xa_ws, mean_ws, rstd_ws);
  k_graph<<<BATCH, 256, 0, stream>>>(xa_ws, h_ws, mean_ws, rstd_ws, gmma, beta, gcnB, outp);
}
